// CrossVit_45148696216621
// MI455X (gfx1250) — matrix operand order repaired and checked by reading, compiled, not run on hardware
//
#include <hip/hip_runtime.h>
#include <hip/hip_bf16.h>
#include <math.h>

// ---------------------------------------------------------------------------
// Single-head attention: out = softmax((x Wq + bq)(x Wk + bk)^T / sqrt(E)) (x Wv + bv)
// SEQ=8192, EMB=2048, fp32 in/out. All GEMMs via v_wmma_f32_16x16x32_f16 with
// double-buffered async LDS staging (GLOBAL_LOAD_ASYNC_TO_LDS_B128 / ASYNCcnt).
// Workspace (f16): xh | Wqt,Wkt,Wvt | Qh,Kh | Vt | P   (~280 MB total)
// ---------------------------------------------------------------------------

#define SEQ 8192
#define EMB 2048

#define KSTEP 32
#define LPAD  8
#define LROW  (KSTEP + LPAD)   // 40 halfs per LDS row: 20*lane mod 64 distinct -> no bank conflicts

typedef _Float16 half8  __attribute__((ext_vector_type(8)));
typedef _Float16 half16 __attribute__((ext_vector_type(16)));
typedef float    float8 __attribute__((ext_vector_type(8)));
typedef int      int4v  __attribute__((ext_vector_type(4)));

typedef __attribute__((address_space(1))) int4v* gptr_b128;
typedef __attribute__((address_space(3))) int4v* lptr_b128;

union H16 { half16 v16; half8 v8[2]; };

#define HAVE_ASYNC_LDS __has_builtin(__builtin_amdgcn_global_load_async_to_lds_b128)

// Stage 32 contiguous bytes (16 halfs) global -> LDS.
__device__ __forceinline__ void stage_tile(const _Float16* __restrict__ gsrc,
                                           _Float16* ldst) {
#if HAVE_ASYNC_LDS
    __builtin_amdgcn_global_load_async_to_lds_b128(
        (gptr_b128)(void*)gsrc, (lptr_b128)(void*)ldst, 0, 0);
    __builtin_amdgcn_global_load_async_to_lds_b128(
        (gptr_b128)(void*)(gsrc + 8), (lptr_b128)(void*)(ldst + 8), 0, 0);
#else
    half8 t0 = *(const half8*)gsrc;
    half8 t1 = *(const half8*)(gsrc + 8);
    *(half8*)ldst       = t0;
    *(half8*)(ldst + 8) = t1;
#endif
}

template <int N>
__device__ __forceinline__ void wait_async() {
#if __has_builtin(__builtin_amdgcn_s_wait_asynccnt)
    __builtin_amdgcn_s_wait_asynccnt(N);
#else
    asm volatile("s_wait_asynccnt %0" :: "n"(N));
#endif
}

// A fragment (16x32 f16, MxK) from LDS tile (row-major, LROW stride):
//  lanes 0-15 : m = lane,    halfs[0..7]=K(0..7),  halfs[8..15]=K(16..23)
//  lanes16-31 : m = lane-16, halfs[0..7]=K(8..15), halfs[8..15]=K(24..31)
__device__ __forceinline__ half16 lds_a_frag(const _Float16* sA, int rowt, int lane) {
    int m    = rowt + (lane & 15);
    int koff = (lane & 16) ? 8 : 0;
    const _Float16* p = sA + m * LROW + koff;
    H16 r;
    r.v8[0] = *(const half8*)(p);
    r.v8[1] = *(const half8*)(p + 16);
    return r.v16;
}

// B fragment (32x16 f16, KxN) from LDS tile of Bsrc[n][k] rows:
//  lanes 0-15 : n = lane,    halfs[0..7]=K(0..7),  halfs[8..15]=K(16..23)
//  lanes16-31 : n = lane-16, halfs[0..7]=K(8..15), halfs[8..15]=K(24..31)
__device__ __forceinline__ half16 lds_b_frag(const _Float16* sB, int colt, int lane) {
    int n  = colt + (lane & 15);
    int kb = (lane & 16) ? 8 : 0;
    const _Float16* p = sB + n * LROW + kb;
    H16 r;
    r.v8[0] = *(const half8*)(p);
    r.v8[1] = *(const half8*)(p + 16);
    return r.v16;
}

__device__ __forceinline__ void compute_step(const _Float16* cA, const _Float16* cB,
                                             int rowt0, int colt0, int lane,
                                             float8 (&acc)[2][4]) {
    half16 a0 = lds_a_frag(cA, rowt0,      lane);
    half16 a1 = lds_a_frag(cA, rowt0 + 16, lane);
    half16 b0 = lds_b_frag(cB, colt0,      lane);
    half16 b1 = lds_b_frag(cB, colt0 + 16, lane);
    half16 b2 = lds_b_frag(cB, colt0 + 32, lane);
    half16 b3 = lds_b_frag(cB, colt0 + 48, lane);
    acc[0][0] = __builtin_amdgcn_wmma_f32_16x16x32_f16(false, a0, false, b0, (short)0, acc[0][0], false, false);
    acc[1][0] = __builtin_amdgcn_wmma_f32_16x16x32_f16(false, a1, false, b0, (short)0, acc[1][0], false, false);
    acc[0][1] = __builtin_amdgcn_wmma_f32_16x16x32_f16(false, a0, false, b1, (short)0, acc[0][1], false, false);
    acc[1][1] = __builtin_amdgcn_wmma_f32_16x16x32_f16(false, a1, false, b1, (short)0, acc[1][1], false, false);
    acc[0][2] = __builtin_amdgcn_wmma_f32_16x16x32_f16(false, a0, false, b2, (short)0, acc[0][2], false, false);
    acc[1][2] = __builtin_amdgcn_wmma_f32_16x16x32_f16(false, a1, false, b2, (short)0, acc[1][2], false, false);
    acc[0][3] = __builtin_amdgcn_wmma_f32_16x16x32_f16(false, a0, false, b3, (short)0, acc[0][3], false, false);
    acc[1][3] = __builtin_amdgcn_wmma_f32_16x16x32_f16(false, a1, false, b3, (short)0, acc[1][3], false, false);
}

// Generic WMMA GEMM: C[M,N] = A[M,K] * Bsrc^T (Bsrc stored [N,K]).
// Block: 256 threads = 8 waves; block tile 128x128; wave tile 32x64 (2x4 frags).
// Per k-step the unique 128x32 A and B tiles are async-DMA'd into double-buffered
// LDS (each thread stages 2x32B); fragments then come from LDS (conflict-free
// ds_load_b128), removing the 2x/4x cross-wave global redundancy.
// mode 0: f16 store Out[m*ldo+n] = (c + bias)*scale
// mode 2: f16 transposed store Out[n*ldo+m] = (c + bias)*scale   (for V^T)
// mode 4: f32 store Out[m*ldo+n] = c
__global__ __launch_bounds__(256)
void gemm_wmma_f16(const _Float16* __restrict__ A,
                   const _Float16* __restrict__ Bsrc,
                   const float* __restrict__ bias,
                   void* __restrict__ Out,
                   int K, int lda, int ldb, int ldo,
                   float scale, int mode) {
    __shared__ __align__(16) _Float16 sA[2][128 * LROW];
    __shared__ __align__(16) _Float16 sB[2][128 * LROW];

    const int tid   = threadIdx.x;
    const int lane  = tid & 31;
    const int wave  = tid >> 5;
    const int rowt0 = (wave & 3) * 32;   // wave's rows within block tile
    const int colt0 = (wave >> 2) * 64;  // wave's cols within block tile
    const int row0  = blockIdx.x * 128;
    const int col0  = blockIdx.y * 128;

    // staging: thread t moves 32B of tile row (t>>1), half-offset (t&1)*16
    const int sr = tid >> 1;
    const int sc = (tid & 1) * 16;
    const _Float16* gA = A    + (size_t)(row0 + sr) * lda + sc;
    const _Float16* gB = Bsrc + (size_t)(col0 + sr) * ldb + sc;
    const int soff = sr * LROW + sc;

    float8 acc[2][4] = {};

    stage_tile(gA, &sA[0][soff]);
    stage_tile(gB, &sB[0][soff]);

    int buf = 0;
    int k0  = 0;
    #pragma unroll 1
    for (; k0 < K - KSTEP; k0 += KSTEP) {
        stage_tile(gA + k0 + KSTEP, &sA[buf ^ 1][soff]);
        stage_tile(gB + k0 + KSTEP, &sB[buf ^ 1][soff]);
        wait_async<4>();   // async loads complete in order: current buffer done
        __syncthreads();
        compute_step(sA[buf], sB[buf], rowt0, colt0, lane, acc);
        __syncthreads();   // all waves done reading before next stage overwrites
        buf ^= 1;
    }
    wait_async<0>();
    __syncthreads();
    compute_step(sA[buf], sB[buf], rowt0, colt0, lane, acc);

    // C/D layout: element r of acc <-> C[rowbase + r + 8*(lane>=16)][col + (lane&15)]
    #pragma unroll
    for (int rt = 0; rt < 2; ++rt) {
        const int mbase = row0 + rowt0 + rt * 16 + ((lane & 16) ? 8 : 0);
        #pragma unroll
        for (int ct = 0; ct < 4; ++ct) {
            const int n  = col0 + colt0 + 16 * ct + (lane & 15);
            const float bv = bias ? bias[n] : 0.0f;
            if (mode == 2) {
                half8 h;
                #pragma unroll
                for (int r = 0; r < 8; ++r)
                    h[r] = (_Float16)((acc[rt][ct][r] + bv) * scale);
                *(half8*)((_Float16*)Out + (size_t)n * ldo + mbase) = h;
            } else if (mode == 4) {
                float* O = (float*)Out;
                #pragma unroll
                for (int r = 0; r < 8; ++r)
                    O[(size_t)(mbase + r) * ldo + n] = acc[rt][ct][r];
            } else {
                _Float16* O = (_Float16*)Out;
                #pragma unroll
                for (int r = 0; r < 8; ++r)
                    O[(size_t)(mbase + r) * ldo + n] =
                        (_Float16)((acc[rt][ct][r] + bv) * scale);
            }
        }
    }
}

__global__ __launch_bounds__(256)
void cvt_f32_to_f16(const float* __restrict__ in, _Float16* __restrict__ out, int n) {
    int i = blockIdx.x * blockDim.x + threadIdx.x;
    if (i < n) out[i] = (_Float16)in[i];
}

// Wt[n*E + k] = W[k*E + n]  (transpose + downconvert; coalesced reads)
__global__ __launch_bounds__(256)
void transpose_to_f16(const float* __restrict__ W, _Float16* __restrict__ Wt, int E) {
    int i = blockIdx.x * blockDim.x + threadIdx.x;
    int k = i / E;
    int n = i - k * E;
    Wt[(size_t)n * E + k] = (_Float16)W[i];
}

// In-place row softmax over P[row][0..SEQ), f16. One 256-thread block per row.
__global__ __launch_bounds__(256)
void softmax_rows(_Float16* __restrict__ P) {
    __shared__ float red[256];
    const int t = threadIdx.x;
    _Float16* p = P + (size_t)blockIdx.x * SEQ;

    float x[SEQ / 256];
    float m = -1e30f;
    #pragma unroll
    for (int j = 0; j < SEQ / 256; ++j) {
        x[j] = (float)p[t + 256 * j];
        m = fmaxf(m, x[j]);
    }
    red[t] = m; __syncthreads();
    for (int s = 128; s > 0; s >>= 1) {
        if (t < s) red[t] = fmaxf(red[t], red[t + s]);
        __syncthreads();
    }
    m = red[0]; __syncthreads();

    float sum = 0.0f;
    #pragma unroll
    for (int j = 0; j < SEQ / 256; ++j) {
        x[j] = __expf(x[j] - m);
        sum += x[j];
    }
    red[t] = sum; __syncthreads();
    for (int s = 128; s > 0; s >>= 1) {
        if (t < s) red[t] += red[t + s];
        __syncthreads();
    }
    const float inv = 1.0f / red[0];
    #pragma unroll
    for (int j = 0; j < SEQ / 256; ++j)
        p[t + 256 * j] = (_Float16)(x[j] * inv);
}

extern "C" void kernel_launch(void* const* d_in, const int* in_sizes, int n_in,
                              void* d_out, int out_size, void* d_ws, size_t ws_size,
                              hipStream_t stream) {
    const float* x  = (const float*)d_in[0];
    const float* Wq = (const float*)d_in[1];
    const float* bq = (const float*)d_in[2];
    const float* Wk = (const float*)d_in[3];
    const float* bk = (const float*)d_in[4];
    const float* Wv = (const float*)d_in[5];
    const float* bv = (const float*)d_in[6];

    char* ws = (char*)d_ws;
    size_t off = 0;
    _Float16* xh  = (_Float16*)(ws + off); off += (size_t)SEQ * EMB * 2;
    _Float16* Wqt = (_Float16*)(ws + off); off += (size_t)EMB * EMB * 2;
    _Float16* Wkt = (_Float16*)(ws + off); off += (size_t)EMB * EMB * 2;
    _Float16* Wvt = (_Float16*)(ws + off); off += (size_t)EMB * EMB * 2;
    _Float16* Qh  = (_Float16*)(ws + off); off += (size_t)SEQ * EMB * 2;
    _Float16* Kh  = (_Float16*)(ws + off); off += (size_t)SEQ * EMB * 2;
    _Float16* Vt  = (_Float16*)(ws + off); off += (size_t)EMB * SEQ * 2;
    _Float16* P   = (_Float16*)(ws + off); off += (size_t)SEQ * SEQ * 2;

    const float scale = 1.0f / sqrtf((float)EMB);

    // 1) downconvert activations + transposed weights
    cvt_f32_to_f16<<<(SEQ * EMB) / 256, 256, 0, stream>>>(x, xh, SEQ * EMB);
    transpose_to_f16<<<(EMB * EMB) / 256, 256, 0, stream>>>(Wq, Wqt, EMB);
    transpose_to_f16<<<(EMB * EMB) / 256, 256, 0, stream>>>(Wk, Wkt, EMB);
    transpose_to_f16<<<(EMB * EMB) / 256, 256, 0, stream>>>(Wv, Wvt, EMB);

    // 2) projections: Q (scale folded), K, V^T
    dim3 gProj(SEQ / 128, EMB / 128);
    gemm_wmma_f16<<<gProj, 256, 0, stream>>>(xh, Wqt, bq, Qh,
                                             EMB, EMB, EMB, EMB, scale, 0);
    gemm_wmma_f16<<<gProj, 256, 0, stream>>>(xh, Wkt, bk, Kh,
                                             EMB, EMB, EMB, EMB, 1.0f, 0);
    gemm_wmma_f16<<<gProj, 256, 0, stream>>>(xh, Wvt, bv, Vt,
                                             EMB, EMB, EMB, SEQ, 1.0f, 2);

    // 3) scores: P = Qh * Kh^T  (Bsrc = Kh stored [t][e] == [N][K])
    dim3 gScore(SEQ / 128, SEQ / 128);
    gemm_wmma_f16<<<gScore, 256, 0, stream>>>(Qh, Kh, nullptr, P,
                                              EMB, EMB, EMB, SEQ, 1.0f, 0);

    // 4) row softmax in place
    softmax_rows<<<SEQ, 256, 0, stream>>>(P);

    // 5) out = P * V  (Bsrc = Vt stored [e][t] == [N][K]); f32 store
    dim3 gOut(SEQ / 128, EMB / 128);
    gemm_wmma_f16<<<gOut, 256, 0, stream>>>(P, Vt, nullptr, d_out,
                                            SEQ, SEQ, SEQ, EMB, 1.0f, 4);
}
